// PosPool_26259430048619
// MI455X (gfx1250) — matrix operand order repaired and checked by reading, not compiled, not run on hardware
//
#include <hip/hip_runtime.h>
#include <hip/hip_bf16.h>

#define N_PTS 20000
#define NSAMP 32
#define C_IN 64
#define C_OUT 128
#define M_ROWS (N_PTS * NSAMP) // 640000

typedef __attribute__((ext_vector_type(16))) __bf16 v16bf;
typedef __attribute__((ext_vector_type(8)))  float  v8f;

#define WMMA_BF16(A, B, C) \
  __builtin_amdgcn_wmma_f32_16x16x32_bf16(false, (A), false, (B), (short)0, (C), false, false)

// A-fragment (16x32 bf16) for v_wmma_f32_16x16x32_bf16, loaded straight from a
// row-major f32 feature row. Lanes 0-15 hold row M=lane, elems
// 0..15 = K[16h..16h+15] (h = lane>=16), matching the contiguous B fragment.
__device__ __forceinline__ v16bf load_a_frag(const float* __restrict__ row, int kc, int half) {
  const float* g0 = row + kc * 32 + half * 16;
  float4 f0 = *(const float4*)(g0);
  float4 f1 = *(const float4*)(g0 + 4);
  float4 f2 = *(const float4*)(g0 + 8);
  float4 f3 = *(const float4*)(g0 + 12);
  v16bf a;
  a[0] = (__bf16)f0.x; a[1] = (__bf16)f0.y; a[2]  = (__bf16)f0.z; a[3]  = (__bf16)f0.w;
  a[4] = (__bf16)f1.x; a[5] = (__bf16)f1.y; a[6]  = (__bf16)f1.z; a[7]  = (__bf16)f1.w;
  a[8] = (__bf16)f2.x; a[9] = (__bf16)f2.y; a[10] = (__bf16)f2.z; a[11] = (__bf16)f2.w;
  a[12] = (__bf16)f3.x; a[13] = (__bf16)f3.y; a[14] = (__bf16)f3.z; a[15] = (__bf16)f3.w;
  return a;
}

__global__ void pospool_ws_init(float* __restrict__ ws) {
  ws[threadIdx.x] = 0.0f; // zero sum[128] + sumsq[128]
}

// Pass 1: per-channel sum and sum-of-squares of x = F * W^T (bf16 WMMA, f32 acc).
__global__ void __launch_bounds__(256) pospool_stats_kernel(
    const float* __restrict__ features, const float* __restrict__ W,
    float* __restrict__ wsSum /* [0..127]=sum, [128..255]=sumsq */) {
  __shared__ __align__(32) __bf16 Wlds[C_OUT * C_IN];
  __shared__ float statsLds[256];
  const int tid = threadIdx.x;
  for (int i = tid; i < C_OUT * C_IN; i += 256) Wlds[i] = (__bf16)W[i];
  statsLds[tid] = 0.0f;
  __syncthreads();

  const int lane = tid & 31, wave = tid >> 5;
  const int half = lane >> 4, col = lane & 15;
  const int waveId = blockIdx.x * 8 + wave;

  float regS[8], regS2[8];
#pragma unroll
  for (int nt = 0; nt < 8; ++nt) { regS[nt] = 0.0f; regS2[nt] = 0.0f; }

  for (int i = 0; i < 8; ++i) {
    const int n = waveId * 8 + i;          // wave-uniform
    if (n >= N_PTS) break;
    const float* f0 = features + ((size_t)n * NSAMP + col) * C_IN;
    const float* f1 = features + ((size_t)n * NSAMP + 16 + col) * C_IN;
    v16bf a00 = load_a_frag(f0, 0, half), a01 = load_a_frag(f0, 1, half);
    v16bf a10 = load_a_frag(f1, 0, half), a11 = load_a_frag(f1, 1, half);
#pragma unroll
    for (int nt = 0; nt < 8; ++nt) {
      const __bf16* wr = &Wlds[(nt * 16 + col) * C_IN + 16 * half];
      v16bf b0 = *(const v16bf*)(wr);        // K = 0..31 slice for this lane
      v16bf b1 = *(const v16bf*)(wr + 32);   // K = 32..63 slice
      v8f acc0 = {}; v8f acc1 = {};
      acc0 = WMMA_BF16(a00, b0, acc0);
      acc0 = WMMA_BF16(a01, b1, acc0);
      acc1 = WMMA_BF16(a10, b0, acc1);
      acc1 = WMMA_BF16(a11, b1, acc1);
      float s = 0.0f, s2 = 0.0f;
#pragma unroll
      for (int v = 0; v < 8; ++v) {
        s  += acc0[v] + acc1[v];
        s2 += acc0[v] * acc0[v] + acc1[v] * acc1[v];
      }
      regS[nt] += s; regS2[nt] += s2;
    }
  }
#pragma unroll
  for (int nt = 0; nt < 8; ++nt) {
    float s  = regS[nt]  + __shfl_xor(regS[nt],  16, 32);
    float s2 = regS2[nt] + __shfl_xor(regS2[nt], 16, 32);
    if (lane < 16) {
      atomicAdd(&statsLds[nt * 16 + col], s);
      atomicAdd(&statsLds[128 + nt * 16 + col], s2);
    }
  }
  __syncthreads();
  atomicAdd(&wsSum[tid], statsLds[tid]);
}

// Tiny kernel: fold BN stats + gamma/beta into per-channel scale/shift.
__global__ void pospool_bn_finalize(const float* __restrict__ wsSum,
                                    const float* __restrict__ gamma,
                                    const float* __restrict__ beta,
                                    float* __restrict__ scaleShift) {
  const int d = threadIdx.x;
  if (d < C_OUT) {
    const float invM = 1.0f / (float)M_ROWS;
    float mean = wsSum[d] * invM;
    float var  = fmaxf(wsSum[128 + d] * invM - mean * mean, 0.0f);
    float sc   = gamma[d] * rsqrtf(var + 1e-5f);
    scaleShift[d]       = sc;
    scaleShift[128 + d] = beta[d] - mean * sc;
  }
}

// Pass 2: GEMM (bf16 WMMA) fused with BN affine + ReLU + geo/mask-weighted pool.
__global__ void __launch_bounds__(256) pospool_main_kernel(
    const float* __restrict__ features, const float* __restrict__ coords,
    const float* __restrict__ centers, const int* __restrict__ mask,
    const float* __restrict__ W, const float* __restrict__ scaleShift,
    float* __restrict__ out) {
  __shared__ __align__(32) __bf16 Wlds[C_OUT * C_IN];
  __shared__ float scLds[C_OUT], shLds[C_OUT];
  __shared__ float wgtLds[8][NSAMP * 4]; // per-wave geo*mask weights
  const int tid = threadIdx.x;
  for (int i = tid; i < C_OUT * C_IN; i += 256) Wlds[i] = (__bf16)W[i];
  if (tid < C_OUT) { scLds[tid] = scaleShift[tid]; shLds[tid] = scaleShift[128 + tid]; }
  __syncthreads();

  const int lane = tid & 31, wave = tid >> 5;
  const int half = lane >> 4, col = lane & 15;
  const int c4 = col & 3;
  const int waveId = blockIdx.x * 8 + wave;

  for (int i = 0; i < 8; ++i) {
    const int n = waveId * 8 + i;          // wave-uniform
    if (n >= N_PTS) break;

    // geometric weights: geo = [rel/0.1, rel/0.1, rel/0.2, dist/0.06] * mask
    const float* crd = coords + ((size_t)n * NSAMP + lane) * 3;
    float cx = centers[n * 4 + 1], cy = centers[n * 4 + 2], cz = centers[n * 4 + 3];
    float rx = crd[0] - cx, ry = crd[1] - cy, rz = crd[2] - cz;
    float dist = rx * rx + ry * ry + rz * rz;
    float m = (float)mask[n * NSAMP + lane];
    float4 wv;
    wv.x = rx * 10.0f * m;
    wv.y = ry * 10.0f * m;
    wv.z = rz * 5.0f * m;
    wv.w = dist * (1.0f / 0.06f) * m;
    *(float4*)&wgtLds[wave][lane * 4] = wv;

    float den = m;
#pragma unroll
    for (int off = 16; off > 0; off >>= 1) den += __shfl_xor(den, off, 32);
    const float invDen = 1.0f / fmaxf(den, 1.0f);

    asm volatile("s_wait_dscnt 0" ::: "memory"); // intra-wave LDS visibility

    float wrow0[8], wrow1[8]; // pool weights for this lane's 8 rows, both M-tiles
#pragma unroll
    for (int v = 0; v < 8; ++v) {
      wrow0[v] = wgtLds[wave][(half * 8 + v) * 4 + c4];
      wrow1[v] = wgtLds[wave][(16 + half * 8 + v) * 4 + c4];
    }

    const float* f0 = features + ((size_t)n * NSAMP + col) * C_IN;
    const float* f1 = features + ((size_t)n * NSAMP + 16 + col) * C_IN;
    v16bf a00 = load_a_frag(f0, 0, half), a01 = load_a_frag(f0, 1, half);
    v16bf a10 = load_a_frag(f1, 0, half), a11 = load_a_frag(f1, 1, half);

#pragma unroll
    for (int nt = 0; nt < 8; ++nt) {
      const __bf16* wr = &Wlds[(nt * 16 + col) * C_IN + 16 * half];
      v16bf b0 = *(const v16bf*)(wr);
      v16bf b1 = *(const v16bf*)(wr + 32);
      const float sc = scLds[nt * 16 + col], sh = shLds[nt * 16 + col];
      v8f acc0 = {}; v8f acc1 = {};
      acc0 = WMMA_BF16(a00, b0, acc0);
      acc0 = WMMA_BF16(a01, b1, acc0);
      acc1 = WMMA_BF16(a10, b0, acc1);
      acc1 = WMMA_BF16(a11, b1, acc1);
      float tot = 0.0f;
#pragma unroll
      for (int v = 0; v < 8; ++v) {
        tot += fmaxf(acc0[v] * sc + sh, 0.0f) * wrow0[v];
        tot += fmaxf(acc1[v] * sc + sh, 0.0f) * wrow1[v];
      }
      tot += __shfl_xor(tot, 16, 32);
      if (lane < 16) out[(size_t)n * C_OUT + nt * 16 + col] = tot * invDen;
    }
  }
}

extern "C" void kernel_launch(void* const* d_in, const int* in_sizes, int n_in,
                              void* d_out, int out_size, void* d_ws, size_t ws_size,
                              hipStream_t stream) {
  (void)in_sizes; (void)n_in; (void)out_size; (void)ws_size;
  const float* features = (const float*)d_in[0];
  const float* coords   = (const float*)d_in[1];
  const float* centers  = (const float*)d_in[2];
  const int*   mask     = (const int*)d_in[3];
  const float* W        = (const float*)d_in[4];
  const float* gamma    = (const float*)d_in[5];
  const float* beta     = (const float*)d_in[6];
  float* out = (float*)d_out;
  float* ws  = (float*)d_ws; // [0..255] sum/sumsq, [256..511] scale/shift

  const int blocks = (N_PTS + 63) / 64; // 64 points per block (8 waves x 8 points)

  pospool_ws_init<<<1, 256, 0, stream>>>(ws);
  pospool_stats_kernel<<<blocks, 256, 0, stream>>>(features, W, ws);
  pospool_bn_finalize<<<1, 128, 0, stream>>>(ws, gamma, beta, ws + 256);
  pospool_main_kernel<<<blocks, 256, 0, stream>>>(features, coords, centers, mask,
                                                  W, ws + 256, out);
}
